// BlockAttention_42588895707688
// MI455X (gfx1250) — matrix operand order repaired and checked by reading, not compiled, not run on hardware
//
#include <hip/hip_runtime.h>

// ---- CDNA5 WMMA types ----
typedef _Float16 h4   __attribute__((ext_vector_type(4)));
typedef _Float16 h8   __attribute__((ext_vector_type(8)));
typedef _Float16 v16h __attribute__((ext_vector_type(16)));
typedef float    v8f  __attribute__((ext_vector_type(8)));
typedef float    f4   __attribute__((ext_vector_type(4)));

constexpr int DIM   = 128;
constexpr int SEQL  = 8192;
constexpr int NBATCH= 8;
constexpr int BSZ   = 256;   // block size
constexpr int HALO  = 255;
constexpr int WIN   = 766;   // bs + 2*halo
constexpr int NBLK  = SEQL / BSZ;          // 32
constexpr int NCH   = 24;    // ceil(768/32) window chunks
constexpr float SCALE = 0.08838834764831845f; // 1/sqrt(128)

// LDS row strides (halves) chosen for bank-conflict avoidance (64 banks x 4B)
constexpr int KSTR = 136;  // k chunk row-major rows (68 dwords; lanes spread 4*r)
constexpr int VSTR = 44;   // v chunk transposed rows (22 dwords; coprime spread)
constexpr int SSTR = 40;   // per-wave S scratch rows (20 dwords; 16B aligned)

static __device__ __forceinline__ v8f splat8(float x) {
  v8f r;
#pragma unroll
  for (int i = 0; i < 8; ++i) r[i] = x;
  return r;
}
static __device__ __forceinline__ h8 zero8h() {
  h8 r;
#pragma unroll
  for (int i = 0; i < 8; ++i) r[i] = (_Float16)0.0f;
  return r;
}
static __device__ __forceinline__ v16h pack16(h8 lo, h8 hi) {
  v16h a;
#pragma unroll
  for (int i = 0; i < 8; ++i) { a[i] = lo[i]; a[8 + i] = hi[i]; }
  return a;
}
static __device__ __forceinline__ v8f wmma16(v16h a, v16h b, v8f c) {
  // v_wmma_f32_16x16x32_f16: (neg_a, A, neg_b, B, c_mod, C, reuse_a, reuse_b)
  return __builtin_amdgcn_wmma_f32_16x16x32_f16(false, a, false, b, (short)0, c,
                                                false, false);
}

// =====================================================================
// Kernel 1: q/k/v = x @ W.T + b   (f32 in, f16 out to workspace)
// grid = B*S/256 blocks of 512 threads (16 waves); wave owns 16 rows.
// =====================================================================
__global__ __launch_bounds__(512) void proj_kernel(
    const float* __restrict__ x,
    const float* __restrict__ Wq, const float* __restrict__ bq,
    const float* __restrict__ Wk, const float* __restrict__ bk,
    const float* __restrict__ Wv, const float* __restrict__ bv,
    _Float16* __restrict__ qws, _Float16* __restrict__ kws,
    _Float16* __restrict__ vws)
{
  __shared__ __attribute__((aligned(16))) _Float16 wlds[DIM * KSTR]; // ~34 KB
  const int tid  = threadIdx.x;
  const int wid  = tid >> 5;
  const int lane = tid & 31;
  const int r    = lane & 15;   // column/row index within 16
  const int hig  = lane >> 4;   // lane-half select
  const long rowbase = (long)blockIdx.x * 256;

  // --- A fragments: 16 rows of x, K=128 split into 4 frags of K=32 ---
  // A layout (16-bit 16x32): lane -> M=r; half h -> K=(h/8)*16 + hig*8 + h%8
  const float* xrow = x + (rowbase + wid * 16 + r) * DIM;
  v16h af[4];
#pragma unroll
  for (int kk = 0; kk < 4; ++kk) {
#pragma unroll
    for (int g = 0; g < 2; ++g) {
      const int base = kk * 32 + g * 16 + hig * 8;
      f4 f0 = *(const f4*)(xrow + base);
      f4 f1 = *(const f4*)(xrow + base + 4);
#pragma unroll
      for (int j = 0; j < 4; ++j) {
        af[kk][g * 8 + j]     = (_Float16)f0[j];
        af[kk][g * 8 + 4 + j] = (_Float16)f1[j];
      }
    }
  }

  const float* Ws[3] = {Wq, Wk, Wv};
  const float* Bs[3] = {bq, bk, bv};
  _Float16*    Os[3] = {qws, kws, vws};

#pragma unroll
  for (int m = 0; m < 3; ++m) {
    // cooperative W (f32) -> LDS (f16), padded row-major [h][d]
    const f4* wsrc = (const f4*)Ws[m];
    for (int i = tid; i < DIM * DIM / 4; i += 512) {
      f4 w = wsrc[i];
      const int row = (i * 4) >> 7;
      const int col = (i * 4) & 127;
#pragma unroll
      for (int j = 0; j < 4; ++j) wlds[row * KSTR + col + j] = (_Float16)w[j];
    }
    __syncthreads();

    const float* bias = Bs[m];
    _Float16*    outp = Os[m];
#pragma unroll
    for (int nt = 0; nt < 8; ++nt) {
      const int n = nt * 16 + r;          // output channel (B-matrix column)
      v8f c = splat8(bias[n]);            // fold bias into accumulator
#pragma unroll
      for (int kk = 0; kk < 4; ++kk) {
        // B layout: lane -> N=r; half h -> K=(h/8)*16 + hig*8 + h%8
        const _Float16* wr = wlds + n * KSTR + kk * 32 + hig * 8;
        v16h bf = pack16(*(const h8*)wr, *(const h8*)(wr + 16));
        c = wmma16(af[kk], bf, c);
      }
      // C layout: lane -> N=r; vgpr v -> M = hig*8 + v
#pragma unroll
      for (int v = 0; v < 8; ++v) {
        const long srow = rowbase + wid * 16 + hig * 8 + v;
        outp[srow * DIM + n] = (_Float16)c[v];
      }
    }
    __syncthreads();  // before wlds is overwritten for next matrix
  }
}

// =====================================================================
// Kernel 2: per (batch, block): out = (q_blk @ k_win^T * scale) @ v_win
// Window streamed in 24 chunks of 32 positions through LDS.
// grid = B*nb = 256 blocks of 512 threads (16 waves).
// =====================================================================
__global__ __launch_bounds__(512) void attn_kernel(
    const _Float16* __restrict__ qws, const _Float16* __restrict__ kws,
    const _Float16* __restrict__ vws, float* __restrict__ out)
{
  __shared__ __attribute__((aligned(16))) _Float16 klds[32 * KSTR];     // ~8.5 KB
  __shared__ __attribute__((aligned(16))) _Float16 vldsT[DIM * VSTR];   // ~11 KB
  __shared__ __attribute__((aligned(16))) _Float16 slds[16 * 16 * SSTR];// 20 KB

  const int tid  = threadIdx.x;
  const int wid  = tid >> 5;
  const int lane = tid & 31;
  const int r    = lane & 15;
  const int hig  = lane >> 4;
  const int bi   = blockIdx.x >> 5;   // batch
  const int ni   = blockIdx.x & 31;   // block index

  // --- persistent Q A-fragments (16 rows x 128) ---
  const _Float16* qrow =
      qws + (long)(bi * SEQL + ni * BSZ + wid * 16 + r) * DIM;
  v16h qf[4];
#pragma unroll
  for (int kk = 0; kk < 4; ++kk) {
    const int b0 = kk * 32 + hig * 8;
    qf[kk] = pack16(*(const h8*)(qrow + b0), *(const h8*)(qrow + b0 + 16));
  }

  v8f o[8];
#pragma unroll
  for (int nt = 0; nt < 8; ++nt) o[nt] = splat8(0.0f);

  const int ldrow = tid >> 4;   // 0..31: chunk row (window position)
  const int ldcg  = tid & 15;   // 16-byte column group

  // register prefetch of a K/V chunk row-slice (zero-fill halo/window tail)
  auto load_chunk = [&](int ch, h8& kf, h8& vf) {
    const int j = ch * 32 + ldrow;            // window position
    const int s = ni * BSZ + j - HALO;        // global sequence index
    kf = zero8h();
    vf = zero8h();
    if (j < WIN && s >= 0 && s < SEQL) {
      const long off = (long)(bi * SEQL + s) * DIM + ldcg * 8;
      kf = *(const h8*)(kws + off);
      vf = *(const h8*)(vws + off);
    }
  };

  h8 kf, vf;
  load_chunk(0, kf, vf);

  for (int ch = 0; ch < NCH; ++ch) {
    // ---- commit prefetched chunk to LDS ----
    if (ch) __syncthreads();                  // prior chunk's readers done
    *(h8*)(klds + ldrow * KSTR + ldcg * 8) = kf;   // K row-major
#pragma unroll
    for (int i = 0; i < 8; ++i)                     // V transposed [n][j]
      vldsT[(ldcg * 8 + i) * VSTR + ldrow] = vf[i];
    __syncthreads();

    // ---- prefetch next chunk into registers (latency hidden by compute) ----
    if (ch + 1 < NCH) load_chunk(ch + 1, kf, vf);

    // ---- S(16x32) = q_strip @ k_chunk^T ----
    v8f sacc[2];
    sacc[0] = splat8(0.0f);
    sacc[1] = splat8(0.0f);
#pragma unroll
    for (int t = 0; t < 2; ++t) {
#pragma unroll
      for (int kk = 0; kk < 4; ++kk) {
        // B element (K=hid, N=j_loc) = k[j_loc][hid]; hid=(h/8)*16 + hig*8 + h%8
        const _Float16* kr = klds + (t * 16 + r) * KSTR + kk * 32 + hig * 8;
        v16h bf = pack16(*(const h8*)kr, *(const h8*)(kr + 16));
        sacc[t] = wmma16(qf[kk], bf, sacc[t]);
      }
    }

    // ---- scale, C-layout -> A-layout transpose through per-wave LDS ----
    _Float16* sw = slds + wid * (16 * SSTR);  // 16 x SSTR halves, wave-private
#pragma unroll
    for (int t = 0; t < 2; ++t)
#pragma unroll
      for (int v = 0; v < 8; ++v)
        sw[(hig * 8 + v) * SSTR + t * 16 + r] = (_Float16)(sacc[t][v] * SCALE);
    // DS ops are in-order per wave; wait for stores before re-reading
    asm volatile("s_wait_dscnt 0" ::: "memory");
    v16h sa;
    {
      const _Float16* sr = sw + r * SSTR;
      sa = pack16(*(const h8*)(sr + hig * 8), *(const h8*)(sr + 16 + hig * 8));
    }

    // ---- out(16x128) += S(16x32) @ v_chunk(32x128) ----
#pragma unroll
    for (int nt = 0; nt < 8; ++nt) {
      // B element (K=j, N=n) = vldsT[n][j]; contiguous in j -> b64 loads
      const _Float16* vr = vldsT + (nt * 16 + r) * VSTR + hig * 8;
      v16h bf;
#pragma unroll
      for (int w = 0; w < 4; ++w) {
        h4 a = *(const h4*)(vr + (w & 1) * 4 + (w >> 1) * 16);
#pragma unroll
        for (int j = 0; j < 4; ++j) bf[w * 4 + j] = a[j];
      }
      o[nt] = wmma16(sa, bf, o[nt]);
    }
  }

  // ---- store f32 output ----
#pragma unroll
  for (int nt = 0; nt < 8; ++nt) {
#pragma unroll
    for (int v = 0; v < 8; ++v) {
      const long srow = (long)bi * SEQL + ni * BSZ + wid * 16 + hig * 8 + v;
      out[srow * DIM + nt * 16 + r] = o[nt][v];
    }
  }
}

// =====================================================================
extern "C" void kernel_launch(void* const* d_in, const int* in_sizes, int n_in,
                              void* d_out, int out_size, void* d_ws,
                              size_t ws_size, hipStream_t stream) {
  const float* x  = (const float*)d_in[0];
  const float* Wq = (const float*)d_in[1];
  const float* bq = (const float*)d_in[2];
  const float* Wk = (const float*)d_in[3];
  const float* bk = (const float*)d_in[4];
  const float* Wv = (const float*)d_in[5];
  const float* bv = (const float*)d_in[6];
  float* out = (float*)d_out;

  // workspace: q/k/v as f16, 16 MB each (48 MB total)
  _Float16* qws = (_Float16*)d_ws;
  _Float16* kws = qws + (size_t)NBATCH * SEQL * DIM;
  _Float16* vws = kws + (size_t)NBATCH * SEQL * DIM;

  proj_kernel<<<NBATCH * SEQL / 256, 512, 0, stream>>>(
      x, Wq, bq, Wk, bk, Wv, bv, qws, kws, vws);
  attn_kernel<<<NBATCH * NBLK, 512, 0, stream>>>(qws, kws, vws, out);
}
